// Re_gptBlock_85907935855135
// MI455X (gfx1250) — matrix operand order repaired and checked by reading, not compiled, not run on hardware
//
#include <hip/hip_runtime.h>
#include <hip/hip_bf16.h>

typedef unsigned short u16;
typedef __attribute__((ext_vector_type(8)))  float   v8f;
typedef __attribute__((ext_vector_type(16))) __bf16  v16bf;
typedef __attribute__((ext_vector_type(8)))  __bf16  v8bf;

// ---------------------------------------------------------------- utilities
__device__ __forceinline__ u16 f2bf(float f) {
  union { float f; unsigned u; } x; x.f = f;
  unsigned r = x.u + 0x7FFFu + ((x.u >> 16) & 1u);   // round-to-nearest-even
  return (u16)(r >> 16);
}

// A-fragment: 16 bf16 as two 8-element runs at p[0..7] and p[16..23]
__device__ __forceinline__ v16bf ld_a(const u16* p) {
  const v8bf lo = *reinterpret_cast<const v8bf*>(p);
  const v8bf hi = *reinterpret_cast<const v8bf*>(p + 16);
  v16bf r;
#pragma unroll
  for (int e = 0; e < 8; ++e) { r[e] = lo[e]; r[e + 8] = hi[e]; }
  return r;
}
// B-fragment: 16 bf16 as two 8-element runs at p[0..7] and p[16..23]
__device__ __forceinline__ v16bf ld_b(const u16* p) {
  const v8bf lo = *reinterpret_cast<const v8bf*>(p);
  const v8bf hi = *reinterpret_cast<const v8bf*>(p + 16);
  v16bf r;
#pragma unroll
  for (int e = 0; e < 8; ++e) { r[e] = lo[e]; r[e + 8] = hi[e]; }
  return r;
}

__device__ __forceinline__ v8f wmma_bf16(v16bf a, v16bf b, v8f c) {
  return __builtin_amdgcn_wmma_f32_16x16x32_bf16(false, a, false, b, (short)0, c,
                                                 false, false);
}

// wave-local DS ordering (cheaper than a block barrier); mnemonic matches
// what the compiler itself emits for gfx1250.
__device__ __forceinline__ void wave_ds_fence() {
  asm volatile("s_wait_dscnt 0x0" ::: "memory");
}

// ---------------------------------------------------------------- generic batched GEMM
// C[M x N] = A[M x K] * Bt[N x K]^T  (Bt stored N-major, i.e. pre-transposed B)
// batch z offsets: off = (z / div) * s1 + (z % div) * s2   (linear: div=1, s2=0)
// LDS double-buffered; global fetch of tile k+1 overlaps WMMAs on tile k;
// global_prefetch_b8 issued two tiles ahead.
template<bool OUT_BF16, bool BIAS, bool GELU_, bool RESID>
__global__ __launch_bounds__(256) void gemm_bf16(
    const u16* __restrict__ A,  int lda, int aDiv, long aS1, long aS2,
    const u16* __restrict__ Bt, int ldb, int bDiv, long bS1, long bS2,
    void* __restrict__ Cv,      int ldc, int cDiv, long cS1, long cS2,
    const float* __restrict__ bias,
    const float* __restrict__ resid, int ldr,
    int M, int N, int K)
{
  __shared__ __align__(16) u16 As[2][128][40];
  __shared__ __align__(16) u16 Bs[2][128][40];
  const int tid = threadIdx.x;
  const int lane = tid & 31, w = tid >> 5;
  const int wr = w & 3, wc = w >> 2;
  const int hlf = (lane >> 4) & 1, l15 = lane & 15;
  const int m0 = blockIdx.x * 128, n0 = blockIdx.y * 128;
  const int z = blockIdx.z;
  A  += (long)(z / aDiv) * aS1 + (long)(z % aDiv) * aS2;
  Bt += (long)(z / bDiv) * bS1 + (long)(z % bDiv) * bS2;
  const long cOff = (long)(z / cDiv) * cS1 + (long)(z % cDiv) * cS2;

  v8f acc[2][4] = {};

  const int lr   = tid >> 1;
  const int lcol = (tid & 1) * 16;
  const int gm = m0 + lr, gn = n0 + lr;
  const bool mok = gm < M, nok = gn < N;
  const u16* Arow = A  + (long)gm * lda + lcol;
  const u16* Brow = Bt + (long)gn * ldb + lcol;

  auto fetch = [&](int k0, uint4& a0, uint4& a1, uint4& b0, uint4& b1) {
    a0 = uint4{0,0,0,0}; a1 = a0; b0 = a0; b1 = a0;
    if (mok) { const uint4* s = (const uint4*)(Arow + k0); a0 = s[0]; a1 = s[1]; }
    if (nok) { const uint4* s = (const uint4*)(Brow + k0); b0 = s[0]; b1 = s[1]; }
  };
  auto stage = [&](int buf, uint4 a0, uint4 a1, uint4 b0, uint4 b1) {
    uint4* da = (uint4*)&As[buf][lr][lcol]; da[0] = a0; da[1] = a1;
    uint4* db = (uint4*)&Bs[buf][lr][lcol]; db[0] = b0; db[1] = b1;
  };

  { // prologue: tile 0 into buffer 0
    uint4 a0, a1, b0, b1;
    fetch(0, a0, a1, b0, b1);
    stage(0, a0, a1, b0, b1);
  }
  __syncthreads();

  int buf = 0;
  for (int k0 = 0; k0 < K; k0 += 32) {
    const int kn = k0 + 32;
    const bool hasNext = kn < K;
    uint4 na0, na1, nb0, nb1;
    if (hasNext) fetch(kn, na0, na1, nb0, nb1);
    if (kn + 32 < K) {                       // gfx1250 global_prefetch_b8
      if (mok) __builtin_prefetch(Arow + kn + 32, 0, 3);
      if (nok) __builtin_prefetch(Brow + kn + 32, 0, 3);
    }

    v16bf af[2], bf_[4];
#pragma unroll
    for (int i = 0; i < 2; ++i)
      af[i] = ld_a(&As[buf][wr * 32 + i * 16 + l15][hlf * 8]);
#pragma unroll
    for (int j = 0; j < 4; ++j)
      bf_[j] = ld_b(&Bs[buf][wc * 64 + j * 16 + l15][hlf * 8]);
#pragma unroll
    for (int i = 0; i < 2; ++i)
#pragma unroll
      for (int j = 0; j < 4; ++j)
        acc[i][j] = wmma_bf16(af[i], bf_[j], acc[i][j]);

    if (hasNext) {
      stage(buf ^ 1, na0, na1, nb0, nb1);
      __syncthreads();
      buf ^= 1;
    }
  }

  // epilogue
#pragma unroll
  for (int j = 0; j < 4; ++j) {
    const int col = n0 + wc * 64 + j * 16 + l15;
    if (col >= N) continue;
    const float bv = BIAS ? bias[col] : 0.0f;
#pragma unroll
    for (int i = 0; i < 2; ++i) {
      const int rbase = m0 + wr * 32 + i * 16 + hlf * 8;
#pragma unroll
      for (int r = 0; r < 8; ++r) {
        const int row = rbase + r;
        if (row >= M) continue;
        float vv = acc[i][j][r] + bv;
        if (GELU_) {
          const float x = vv;
          vv = 0.5f * x * (1.0f + tanhf(0.7978845608028654f *
                                        (x + 0.044715f * x * x * x)));
        }
        if (RESID) vv += resid[(long)row * ldr + col];
        const long idx = cOff + (long)row * ldc + col;
        if (OUT_BF16) ((u16*)Cv)[idx] = f2bf(vv);
        else          ((float*)Cv)[idx] = vv;
      }
    }
  }
}

// ---------------------------------------------------------------- flash self-attention
// q,k,v,o : bf16 [b][n][H*HD], head h occupies cols h*HD..h*HD+127. No QK scale;
// causal with bias -1e9 (matches reference). 4 waves, 16 query rows per wave.
// KV tiles double-buffered in LDS; P relayout uses wave-local s_wait_dscnt.
__global__ __launch_bounds__(128) void flash_self_attn(
    const u16* __restrict__ q, const u16* __restrict__ k,
    const u16* __restrict__ v, u16* __restrict__ o)
{
  const int n = 1024, D = 2048, HD = 128;
  const int bh = blockIdx.x;            // b*16 + h
  const int b = bh >> 4, h = bh & 15;
  const int qt = blockIdx.y;            // 64-row q tile
  const long base = (long)b * n * D + (long)h * HD;
  const int tid = threadIdx.x, lane = tid & 31, w = tid >> 5;
  const int hlf = (lane >> 4) & 1, l15 = lane & 15;
  const int r0 = qt * 64 + w * 16;

  __shared__ __align__(16) u16 Ks[2][32][136];   // [j][d]  (Bt layout for QK^T)
  __shared__ __align__(16) u16 Vt[2][128][40];   // [d][j]  (Bt layout for P*V)
  __shared__ __align__(16) u16 Ps[4][16][40];    // per-wave P in A-tile layout

  v16bf qf[4];
#pragma unroll
  for (int ks = 0; ks < 4; ++ks)
    qf[ks] = ld_a(q + base + (long)(r0 + l15) * D + ks * 32 + hlf * 8);

  v8f oacc[8] = {};
  float mrow[8], lrow[8];
#pragma unroll
  for (int r = 0; r < 8; ++r) { mrow[r] = -3.0e38f; lrow[r] = 0.0f; }

  const int kr = tid >> 2, kc = (tid & 3) * 32;  // K-tile fill coords
  uint4 kreg[4]; v8bf vreg[4];
  auto fetchKV = [&](int j0) {
    const uint4* s = (const uint4*)(k + base + (long)(j0 + kr) * D + kc);
    kreg[0] = s[0]; kreg[1] = s[1]; kreg[2] = s[2]; kreg[3] = s[3];
#pragma unroll
    for (int u = 0; u < 4; ++u) {
      const int chunk = tid * 4 + u;              // 512 chunks of 8
      const int jr = chunk >> 4, dc = (chunk & 15) * 8;
      vreg[u] = *(const v8bf*)(v + base + (long)(j0 + jr) * D + dc);
    }
  };
  auto stageKV = [&](int buf) {
    uint4* d = (uint4*)&Ks[buf][kr][kc];
    d[0] = kreg[0]; d[1] = kreg[1]; d[2] = kreg[2]; d[3] = kreg[3];
#pragma unroll
    for (int u = 0; u < 4; ++u) {
      const int chunk = tid * 4 + u;
      const int jr = chunk >> 4, dc = (chunk & 15) * 8;
      const u16* pv = (const u16*)&vreg[u];
#pragma unroll
      for (int e = 0; e < 8; ++e) Vt[buf][dc + e][jr] = pv[e];
    }
  };

  const int ntiles = 2 * qt + 2;                 // causal bound: j <= qt*64+63
  fetchKV(0); stageKV(0);
  __syncthreads();

  int buf = 0;
  for (int jt = 0; jt < ntiles; ++jt) {
    const int j0 = jt * 32;
    const bool hasNext = (jt + 1) < ntiles;
    if (hasNext) fetchKV(j0 + 32);
    if (jt + 2 < ntiles) {                       // gfx1250 global_prefetch_b8
      __builtin_prefetch(k + base + (long)(j0 + 64 + kr) * D + kc, 0, 3);
      __builtin_prefetch(v + base + (long)(j0 + 64 + kr) * D + kc, 0, 3);
    }

    // S = Q * K^T   (16 x 32 per wave)
    v8f sacc[2] = {};
#pragma unroll
    for (int ks = 0; ks < 4; ++ks) {
#pragma unroll
      for (int sub = 0; sub < 2; ++sub) {
        v16bf bb = ld_b(&Ks[buf][sub * 16 + l15][ks * 32 + hlf * 8]);
        sacc[sub] = wmma_bf16(qf[ks], bb, sacc[sub]);
      }
    }

    // online softmax per register-row (row lives across a 16-lane half)
#pragma unroll
    for (int r = 0; r < 8; ++r) {
      const int row = r0 + hlf * 8 + r;
      float s0 = sacc[0][r]; if (j0 + l15 > row)      s0 = -1.0e9f;
      float s1 = sacc[1][r]; if (j0 + 16 + l15 > row) s1 = -1.0e9f;
      float mx = fmaxf(s0, s1);
#pragma unroll
      for (int d = 1; d < 16; d <<= 1) mx = fmaxf(mx, __shfl_xor(mx, d, 32));
      const float mnew  = fmaxf(mrow[r], mx);
      const float scale = __expf(mrow[r] - mnew);
      const float p0 = __expf(s0 - mnew);
      const float p1 = __expf(s1 - mnew);
      float rs = p0 + p1;
#pragma unroll
      for (int d = 1; d < 16; d <<= 1) rs += __shfl_xor(rs, d, 32);
      lrow[r] = lrow[r] * scale + rs;
      mrow[r] = mnew;
#pragma unroll
      for (int d8 = 0; d8 < 8; ++d8) oacc[d8][r] *= scale;
      Ps[w][hlf * 8 + r][l15]      = f2bf(p0);
      Ps[w][hlf * 8 + r][16 + l15] = f2bf(p1);
    }
    wave_ds_fence();                              // Ps is wave-private

    // O += P * V
    const v16bf pf = ld_a(&Ps[w][l15][hlf * 8]);
#pragma unroll
    for (int d8 = 0; d8 < 8; ++d8) {
      v16bf vb = ld_b(&Vt[buf][d8 * 16 + l15][hlf * 8]);
      oacc[d8] = wmma_bf16(pf, vb, oacc[d8]);
    }

    if (hasNext) {
      stageKV(buf ^ 1);
      __syncthreads();
      buf ^= 1;
    }
  }

#pragma unroll
  for (int d8 = 0; d8 < 8; ++d8)
#pragma unroll
    for (int r = 0; r < 8; ++r) {
      const int row = r0 + hlf * 8 + r;
      o[base + (long)row * D + d8 * 16 + l15] = f2bf(oacc[d8][r] / lrow[r]);
    }
}

// ---------------------------------------------------------------- elementwise kernels
__global__ __launch_bounds__(256) void layernorm_bf16(
    const float* __restrict__ x, const float* __restrict__ g,
    const float* __restrict__ bb, u16* __restrict__ out, int D)
{
  const int row = blockIdx.x;
  const float* xr = x + (long)row * D;
  __shared__ float red0[8], red1[8];
  float s = 0.f, ss = 0.f;
  for (int c = threadIdx.x; c < D; c += 256) { float v = xr[c]; s += v; ss += v * v; }
#pragma unroll
  for (int d = 1; d < 32; d <<= 1) { s += __shfl_xor(s, d, 32); ss += __shfl_xor(ss, d, 32); }
  const int w = threadIdx.x >> 5, lane = threadIdx.x & 31;
  if (lane == 0) { red0[w] = s; red1[w] = ss; }
  __syncthreads();
  float fs = 0.f, fss = 0.f;
#pragma unroll
  for (int i = 0; i < 8; ++i) { fs += red0[i]; fss += red1[i]; }
  const float mean = fs / D;
  const float var  = fss / D - mean * mean;
  const float inv  = rsqrtf(var + 1e-5f);
  for (int c = threadIdx.x; c < D; c += 256)
    out[(long)row * D + c] = f2bf((xr[c] - mean) * inv * g[c] + bb[c]);
}

// out row (b,i) = rmsnorm(x[b, i+shift]) or 0 past the end   (fused CCA query prep)
__global__ __launch_bounds__(256) void rms_shift_bf16(
    const float* __restrict__ x, const float* __restrict__ g,
    u16* __restrict__ out, int n, int D, int shift)
{
  const int row = blockIdx.x;
  const int b = row / n, i = row % n;
  const int src = i + shift;
  u16* orow = out + (long)row * D;
  if (src >= n) {
    for (int c = threadIdx.x; c < D; c += 256) orow[c] = 0;
    return;
  }
  const float* xr = x + ((long)b * n + src) * D;
  __shared__ float red[8];
  float ss = 0.f;
  for (int c = threadIdx.x; c < D; c += 256) { float v = xr[c]; ss += v * v; }
#pragma unroll
  for (int d = 1; d < 32; d <<= 1) ss += __shfl_xor(ss, d, 32);
  const int w = threadIdx.x >> 5, lane = threadIdx.x & 31;
  if (lane == 0) red[w] = ss;
  __syncthreads();
  float fss = 0.f;
#pragma unroll
  for (int i2 = 0; i2 < 8; ++i2) fss += red[i2];
  const float norm = sqrtf(fss / D);
  const float inv  = 1.0f / fmaxf(norm, 1e-8f);
  for (int c = threadIdx.x; c < D; c += 256)
    orow[c] = f2bf(xr[c] * inv * g[c]);
}

__global__ void cvt_f32_bf16(const float* __restrict__ in, u16* __restrict__ out,
                             long nElem)
{
  const long i = (long)blockIdx.x * 256 + threadIdx.x;
  if (i < nElem) out[i] = f2bf(in[i]);
}

// Wt[n][k] = bf16(W[k][n])
__global__ __launch_bounds__(256) void transpose_cvt(
    const float* __restrict__ W, u16* __restrict__ Wt, int K, int N)
{
  __shared__ float tile[32][33];
  const int k0 = blockIdx.x * 32, n0 = blockIdx.y * 32;
  const int tx = threadIdx.x & 31, ty = threadIdx.x >> 5;
  for (int r = ty; r < 32; r += 8) {
    const int kk = k0 + r, nn = n0 + tx;
    tile[r][tx] = (kk < K && nn < N) ? W[(long)kk * N + nn] : 0.0f;
  }
  __syncthreads();
  for (int r = ty; r < 32; r += 8) {
    const int nn = n0 + r, kk = k0 + tx;
    if (nn < N && kk < K) Wt[(long)nn * K + kk] = f2bf(tile[tx][r]);
  }
}

// cq fp32 [2048][1024] -> qhat bf16 [bk*16+h][64][64], scale 1/8, RoPE(rot=32)
__global__ void rope_q_kernel(const float* __restrict__ cq,
                              const float* __restrict__ qpe,
                              u16* __restrict__ qhat)
{
  const long idx = (long)blockIdx.x * 256 + threadIdx.x;
  if (idx >= 2048L * 1024) return;
  const int col = (int)(idx & 1023), row = (int)(idx >> 10);
  const int h = col >> 6, d = col & 63;
  const int p = row & 63, c = (row >> 6) & 15, b = row >> 10;
  const float scale = 0.125f;              // 64^-0.5
  const float t = scale * cq[idx];
  float outv;
  if (d < 32) {
    const float f = (p == 0) ? qpe[63 * 32 + d] : 0.0f;
    const float rot = (d < 16)
        ? -scale * cq[(long)row * 1024 + h * 64 + d + 16]
        :  scale * cq[(long)row * 1024 + h * 64 + d - 16];
    outv = t * __cosf(f) + rot * __sinf(f);
  } else outv = t;
  const int bk = b * 16 + c;
  qhat[(((long)bk * 16 + h) * 64 + p) * 64 + d] = f2bf(outv);
}

// ck fp32 [8192][1024] -> khat bf16 [bk*16+h][288][64] at slot j+1, RoPE
__global__ void rope_k_kernel(const float* __restrict__ ck,
                              const float* __restrict__ kpe,
                              u16* __restrict__ khat)
{
  const long idx = (long)blockIdx.x * 256 + threadIdx.x;
  if (idx >= 8192L * 1024) return;
  const int col = (int)(idx & 1023), row = (int)(idx >> 10);
  const int h = col >> 6, d = col & 63;
  const int j = row & 255, bk = row >> 8;
  const float t = ck[idx];
  float outv;
  if (d < 32) {
    const float f = kpe[(j & 127) * 32 + d];
    const float rot = (d < 16)
        ? -ck[(long)row * 1024 + h * 64 + d + 16]
        :  ck[(long)row * 1024 + h * 64 + d - 16];
    outv = t * __cosf(f) + rot * __sinf(f);
  } else outv = t;
  khat[(((long)bk * 16 + h) * 288 + 1 + j) * 64 + d] = f2bf(outv);
}

// khat: slot 0 = null_k head slice, slots 257..287 = 0
__global__ void fill_null_pad_k(const float* __restrict__ nullk,
                                u16* __restrict__ khat)
{
  const long idx = (long)blockIdx.x * 256 + threadIdx.x;
  if (idx >= 512L * 32 * 64) return;
  const int d = (int)(idx & 63);
  const int js = (int)((idx >> 6) & 31);
  const int bkh = (int)(idx >> 11);
  const int j = (js == 0) ? 0 : 256 + js;   // 0 or 257..287
  const int h = bkh & 15;
  const u16 val = (js == 0) ? f2bf(nullk[h * 64 + d]) : (u16)0;
  khat[((long)bkh * 288 + j) * 64 + d] = val;
}

// vt[bkh][d][j] (Bt layout for P*V):  j=0 null_v, 1..256 from cv(fp32), 257..287 zero
__global__ void assemble_vhat(const float* __restrict__ cv,
                              const float* __restrict__ nullv,
                              u16* __restrict__ vt)
{
  const long idx = (long)blockIdx.x * 256 + threadIdx.x;
  if (idx >= 512L * 64 * 288) return;
  const int j = (int)(idx % 288);
  const long t = idx / 288;
  const int d = (int)(t & 63);
  const int bkh = (int)(t >> 6);
  const int h = bkh & 15, bk = bkh >> 4;
  u16 val;
  if (j == 0)        val = f2bf(nullv[h * 64 + d]);
  else if (j <= 256) val = f2bf(cv[((long)bk * 256 + (j - 1)) * 1024 + h * 64 + d]);
  else               val = 0;
  vt[idx] = val;
}

// softmax over 257 valid cols of 288; writes bf16 probs (zeros in the pad)
__global__ __launch_bounds__(256) void softmax_cca(
    const float* __restrict__ S, u16* __restrict__ P)
{
  const int row = blockIdx.x * 8 + (threadIdx.x >> 5);
  const int lane = threadIdx.x & 31;
  const float* sr = S + (long)row * 288;
  u16* pr = P + (long)row * 288;
  float mx = -3.0e38f;
  for (int c = lane; c < 257; c += 32) mx = fmaxf(mx, sr[c]);
#pragma unroll
  for (int d = 1; d < 32; d <<= 1) mx = fmaxf(mx, __shfl_xor(mx, d, 32));
  float sum = 0.f;
  for (int c = lane; c < 257; c += 32) sum += __expf(sr[c] - mx);
#pragma unroll
  for (int d = 1; d < 32; d <<= 1) sum += __shfl_xor(sum, d, 32);
  const float inv = 1.0f / sum;
  for (int c = lane; c < 288; c += 32)
    pr[c] = (c < 257) ? f2bf(__expf(sr[c] - mx) * inv) : (u16)0;
}

// x2 = x1 + shift_right(cca, 63)
__global__ void shift_add(const float* __restrict__ x1,
                          const float* __restrict__ cca,
                          float* __restrict__ x2, int n, int D, int shift)
{
  const long idx = (long)blockIdx.x * 256 + threadIdx.x;
  if (idx >= 2L * n * D) return;
  const int c = (int)(idx % D);
  const long rr = idx / D;
  const int t = (int)(rr % n);
  const int b = (int)(rr / n);
  const float add = (t >= shift) ? cca[(((long)b * n) + (t - shift)) * D + c] : 0.0f;
  x2[idx] = x1[idx] + add;
}

// ---------------------------------------------------------------- host
extern "C" void kernel_launch(void* const* d_in, const int* in_sizes, int n_in,
                              void* d_out, int out_size, void* d_ws, size_t ws_size,
                              hipStream_t stream) {
  (void)in_sizes; (void)n_in; (void)out_size; (void)ws_size;
  const float* x0    = (const float*)d_in[0];
  const float* retr  = (const float*)d_in[1];
  const float* qpe   = (const float*)d_in[2];
  const float* kpe   = (const float*)d_in[3];
  const float* ln1g  = (const float*)d_in[4];
  const float* ln1b  = (const float*)d_in[5];
  const float* Wq    = (const float*)d_in[6];
  const float* Wk    = (const float*)d_in[7];
  const float* Wv    = (const float*)d_in[8];
  const float* Wo    = (const float*)d_in[9];
  const float* bo    = (const float*)d_in[10];
  const float* ccag  = (const float*)d_in[11];
  const float* cWq   = (const float*)d_in[12];
  const float* cWk   = (const float*)d_in[13];
  const float* cWv   = (const float*)d_in[14];
  const float* cWo   = (const float*)d_in[15];
  const float* cbo   = (const float*)d_in[16];
  const float* nullk = (const float*)d_in[17];
  const float* nullv = (const float*)d_in[18];
  const float* ln2g  = (const float*)d_in[19];
  const float* ln2b  = (const float*)d_in[20];
  const float* Wfc   = (const float*)d_in[21];
  const float* bfc   = (const float*)d_in[22];
  const float* Wproj = (const float*)d_in[23];
  const float* bproj = (const float*)d_in[24];
  float* out = (float*)d_out;

  char* wp = (char*)d_ws;
  auto bump = [&](size_t bytes) -> void* {
    void* r = (void*)wp; wp += (bytes + 255) & ~(size_t)255; return r;
  };

  // bf16 transposed weights [N][K]  (WqT/WkT/WvT contiguous; cWkT/cWvT contiguous)
  u16* WqT    = (u16*)bump(2048L * 2048 * 2);
  u16* WkT    = (u16*)bump(2048L * 2048 * 2);
  u16* WvT    = (u16*)bump(2048L * 2048 * 2);
  u16* WoT    = (u16*)bump(2048L * 2048 * 2);
  u16* cWqT   = (u16*)bump(1024L * 2048 * 2);
  u16* cWkT   = (u16*)bump(1024L * 2048 * 2);
  u16* cWvT   = (u16*)bump(1024L * 2048 * 2);
  u16* cWoT   = (u16*)bump(2048L * 1024 * 2);
  u16* WfcT   = (u16*)bump(8192L * 2048 * 2);
  u16* WprojT = (u16*)bump(2048L * 8192 * 2);
  // activations (qb/kb/vb contiguous; ck/cvf contiguous)
  u16*   xln    = (u16*)bump(2048L * 2048 * 2);
  u16*   qb     = (u16*)bump(2048L * 2048 * 2);
  u16*   kb     = (u16*)bump(2048L * 2048 * 2);
  u16*   vb     = (u16*)bump(2048L * 2048 * 2);
  u16*   attnb  = (u16*)bump(2048L * 2048 * 2);
  float* x1     = (float*)bump(2048L * 2048 * 4);
  u16*   xq     = (u16*)bump(2048L * 2048 * 2);
  u16*   ctxb   = (u16*)bump(8192L * 2048 * 2);
  float* cq     = (float*)bump(2048L * 1024 * 4);
  float* ck     = (float*)bump(8192L * 1024 * 4);
  float* cvf    = (float*)bump(8192L * 1024 * 4);
  u16*   qhat   = (u16*)bump(512L * 64 * 64 * 2);
  u16*   khat   = (u16*)bump(512L * 288 * 64 * 2);
  u16*   vhatt  = (u16*)bump(512L * 64 * 288 * 2);
  float* Sbuf   = (float*)bump(512L * 64 * 288 * 4);
  u16*   Pbuf   = (u16*)bump(512L * 64 * 288 * 2);
  u16*   coutb  = (u16*)bump(2048L * 1024 * 2);
  float* ccaraw = (float*)bump(2048L * 2048 * 4);
  float* x2     = (float*)bump(2048L * 2048 * 4);
  u16*   hln    = (u16*)bump(2048L * 2048 * 2);
  u16*   a1     = (u16*)bump(2048L * 8192 * 2);
  (void)kb; (void)vb; (void)cvf;

  const dim3 B256(256), B128(128);

  // ---- weight prep
  transpose_cvt<<<dim3( 64,  64), B256, 0, stream>>>(Wq,    WqT,    2048, 2048);
  transpose_cvt<<<dim3( 64,  64), B256, 0, stream>>>(Wk,    WkT,    2048, 2048);
  transpose_cvt<<<dim3( 64,  64), B256, 0, stream>>>(Wv,    WvT,    2048, 2048);
  transpose_cvt<<<dim3( 64,  64), B256, 0, stream>>>(Wo,    WoT,    2048, 2048);
  transpose_cvt<<<dim3( 64,  32), B256, 0, stream>>>(cWq,   cWqT,   2048, 1024);
  transpose_cvt<<<dim3( 64,  32), B256, 0, stream>>>(cWk,   cWkT,   2048, 1024);
  transpose_cvt<<<dim3( 64,  32), B256, 0, stream>>>(cWv,   cWvT,   2048, 1024);
  transpose_cvt<<<dim3( 32,  64), B256, 0, stream>>>(cWo,   cWoT,   1024, 2048);
  transpose_cvt<<<dim3( 64, 256), B256, 0, stream>>>(Wfc,   WfcT,   2048, 8192);
  transpose_cvt<<<dim3(256,  64), B256, 0, stream>>>(Wproj, WprojT, 8192, 2048);

  // ---- self-attention
  layernorm_bf16<<<2048, B256, 0, stream>>>(x0, ln1g, ln1b, xln, 2048);
  // fused QKV: batch z over the 3 contiguous weight/output buffers
  gemm_bf16<true, false, false, false><<<dim3(16, 16, 3), B256, 0, stream>>>(
      xln, 2048, 1, 0, 0, WqT, 2048, 1, 2048L * 2048, 0,
      qb, 2048, 1, 2048L * 2048, 0, nullptr, nullptr, 0, 2048, 2048, 2048);
  flash_self_attn<<<dim3(32, 16), B128, 0, stream>>>(qb, kb, vb, attnb);
  gemm_bf16<false, true, false, true><<<dim3(16, 16, 1), B256, 0, stream>>>(
      attnb, 2048, 1, 0, 0, WoT, 2048, 1, 0, 0, x1, 2048, 1, 0, 0,
      bo, x0, 2048, 2048, 2048, 2048);

  // ---- chunked cross-attention
  rms_shift_bf16<<<2048, B256, 0, stream>>>(x1, ccag, xq, 1024, 2048, 63);
  cvt_f32_bf16<<<65536, B256, 0, stream>>>(retr, ctxb, 8192L * 2048);
  gemm_bf16<false, false, false, false><<<dim3(16, 8, 1), B256, 0, stream>>>(
      xq, 2048, 1, 0, 0, cWqT, 2048, 1, 0, 0, cq, 1024, 1, 0, 0,
      nullptr, nullptr, 0, 2048, 1024, 2048);
  // fused ck/cv: batch z over contiguous cWkT/cWvT and contiguous ck/cvf
  gemm_bf16<false, false, false, false><<<dim3(64, 8, 2), B256, 0, stream>>>(
      ctxb, 2048, 1, 0, 0, cWkT, 2048, 1, 1024L * 2048, 0,
      ck, 1024, 1, 8192L * 1024, 0, nullptr, nullptr, 0, 8192, 1024, 2048);
  rope_q_kernel<<<8192, B256, 0, stream>>>(cq, qpe, qhat);
  rope_k_kernel<<<32768, B256, 0, stream>>>(ck, kpe, khat);
  fill_null_pad_k<<<4096, B256, 0, stream>>>(nullk, khat);
  assemble_vhat<<<36864, B256, 0, stream>>>(cvf, nullv, vhatt);
  // sim = qhat * khat^T : batch 512, M=64, N=288 (257 valid), K=64
  gemm_bf16<false, false, false, false><<<dim3(1, 3, 512), B256, 0, stream>>>(
      qhat, 64, 1, 64L * 64, 0, khat, 64, 1, 288L * 64, 0,
      Sbuf, 288, 1, 64L * 288, 0, nullptr, nullptr, 0, 64, 288, 64);
  softmax_cca<<<4096, B256, 0, stream>>>(Sbuf, Pbuf);
  // out = P * V : batch 512, M=64, N=64, K=288; scatter back to [rows][1024]
  gemm_bf16<true, false, false, false><<<dim3(1, 1, 512), B256, 0, stream>>>(
      Pbuf, 288, 1, 64L * 288, 0, vhatt, 288, 1, 64L * 288, 0,
      coutb, 1024, 16, 64L * 1024, 64, nullptr, nullptr, 0, 64, 64, 288);
  gemm_bf16<false, true, false, false><<<dim3(16, 16, 1), B256, 0, stream>>>(
      coutb, 1024, 1, 0, 0, cWoT, 1024, 1, 0, 0, ccaraw, 2048, 1, 0, 0,
      cbo, nullptr, 0, 2048, 2048, 1024);
  shift_add<<<16384, B256, 0, stream>>>(x1, ccaraw, x2, 1024, 2048, 63);

  // ---- MLP
  layernorm_bf16<<<2048, B256, 0, stream>>>(x2, ln2g, ln2b, hln, 2048);
  gemm_bf16<true, true, true, false><<<dim3(16, 64, 1), B256, 0, stream>>>(
      hln, 2048, 1, 0, 0, WfcT, 2048, 1, 0, 0, a1, 8192, 1, 0, 0,
      bfc, nullptr, 0, 2048, 8192, 2048);
  gemm_bf16<false, true, false, true><<<dim3(16, 16, 1), B256, 0, stream>>>(
      a1, 8192, 1, 0, 0, WprojT, 8192, 1, 0, 0, out, 2048, 1, 0, 0,
      bproj, x2, 2048, 2048, 2048, 8192);
}
